// GRUCell_21612275433682
// MI455X (gfx1250) — matrix operand order repaired and checked by reading, not compiled, not run on hardware
//
#include <hip/hip_runtime.h>

// ---------------------------------------------------------------------------
// Fused SNN-GRU(+LIF) for MI455X / gfx1250 (wave32, WMMA, TDM).
// Pass 1: convert X [B,I,T] f32 -> Xb [B*T, I] bf16 (pre-transposed, GEMM-ready)
//         convert W_ih [3H, I] f32 -> Wb bf16              (both in d_ws)
// Pass 2: bf16 WMMA GEMM, tiles DMA'd to LDS by the Tensor Data Mover
//         (double-buffered, TENSORcnt), fused bias+heaviside gates + 4-step
//         LIF epilogue entirely in registers -> spikes [B,H,T].
// W_hh is mathematically dead (h0 == 0 -> only b_hh survives).
// ---------------------------------------------------------------------------

typedef __attribute__((ext_vector_type(16))) __bf16 v16bf;
typedef __attribute__((ext_vector_type(8)))  __bf16 v8bf;
typedef __attribute__((ext_vector_type(8)))  float  v8f;
typedef __attribute__((ext_vector_type(4)))  unsigned int u32x4;
typedef __attribute__((ext_vector_type(8)))  int i32x8;
typedef __attribute__((ext_vector_type(4)))  int i32x4;

#define B_   2048
#define I_   2048
#define H_   2048
#define T_   4
#define M_   (B_ * T_)   // 8192 GEMM rows
#define BK   32          // K per WMMA step (bf16)
#define TM   128         // rows per workgroup (= 32 batches)
#define TN   64          // h-columns per workgroup (per gate; x3 gates)
#define LDA  40          // padded LDS row stride in bf16 elems (80B; TDM pad 16DW->+4DW)

#define XB_BYTES ((size_t)M_ * I_ * 2)            // 33,554,432
#define WB_BYTES ((size_t)3 * H_ * I_ * 2)        // 25,165,824
#define WS_NEED  (XB_BYTES + WB_BYTES)

#if defined(__has_builtin)
#if __has_builtin(__builtin_amdgcn_tensor_load_to_lds) && \
    __has_builtin(__builtin_amdgcn_s_wait_tensorcnt)
#define HAVE_TDM 1
#endif
#endif

__device__ __forceinline__ unsigned short f2bf(float f) {
  union { float f; unsigned u; } v; v.f = f;
  unsigned u = v.u;
  u += 0x7FFFu + ((u >> 16) & 1u);   // round-to-nearest-even
  return (unsigned short)(u >> 16);
}

union Frag { v16bf v; struct { v8bf lo, hi; } h; };

// ---------------------------------------------------------------------------
// Pre-pass: f32 -> bf16 conversions (run once per launch; ~7us of streaming)
// ---------------------------------------------------------------------------
__global__ __launch_bounds__(256)
void cvt_x_bf16(const float* __restrict__ X, unsigned short* __restrict__ Xb) {
  // X[b, k, t] -> Xb[b*T+t, k]; thread = (b, k-quad): 16 contiguous floats in,
  // four coalesced 8B rows out.
  const unsigned idx = blockIdx.x * 256 + threadIdx.x;   // B * I/4 threads
  const unsigned b   = idx >> 9;                         // I/4 = 512 quads
  const unsigned kq  = idx & 511;
  float v[16];
  const float4* s = (const float4*)(X + (size_t)b * (I_ * T_) + (size_t)kq * 16);
  *(float4*)(v + 0)  = s[0]; *(float4*)(v + 4)  = s[1];
  *(float4*)(v + 8)  = s[2]; *(float4*)(v + 12) = s[3];
#pragma unroll
  for (int t = 0; t < 4; ++t) {
    const unsigned h0 = f2bf(v[0 * 4 + t]), h1 = f2bf(v[1 * 4 + t]);
    const unsigned h2 = f2bf(v[2 * 4 + t]), h3 = f2bf(v[3 * 4 + t]);
    *(uint2*)&Xb[(size_t)(b * T_ + t) * I_ + kq * 4] =
        make_uint2(h0 | (h1 << 16), h2 | (h3 << 16));
  }
}

__global__ __launch_bounds__(256)
void cvt_w_bf16(const float* __restrict__ W, unsigned short* __restrict__ Wb) {
  const unsigned idx = blockIdx.x * 256 + threadIdx.x;   // 3H*I/4 threads
  const float4 q = ((const float4*)W)[idx];
  const unsigned h0 = f2bf(q.x), h1 = f2bf(q.y), h2 = f2bf(q.z), h3 = f2bf(q.w);
  ((uint2*)Wb)[idx] = make_uint2(h0 | (h1 << 16), h2 | (h3 << 16));
}

// ---------------------------------------------------------------------------
// Tensor Data Mover: one D# per tile. Pads +4 DWORDs after every 16 DWORDs
// (64B row) -> LDS row stride 80B == LDA. B uses a 3D view [gate, n, k].
// ---------------------------------------------------------------------------
__device__ __forceinline__ void tdm_issue(unsigned lds, unsigned long long ga,
                                          unsigned tdim1, int tile1w4,
                                          unsigned long long d1s, int tdim2) {
#if defined(HAVE_TDM)
  u32x4 g0 = { 1u,                                   // count=1 (valid, user mode)
               lds,                                  // lds_addr
               (unsigned)ga,                         // global_addr[31:0]
               ((unsigned)((ga >> 32) & 0x01FFFFFFull)) | (2u << 30) }; // [56:32]|type=2
  i32x8 g1 = { (int)((1u << 16) | (1u << 20) | (3u << 22) | (3u << 25)),
               //     data=2B     pad_en       every 16DW    +4DW
               (int)(((unsigned)I_ & 0xFFFFu) << 16),        // tensor_dim0 lo16
               (int)((((unsigned)I_) >> 16) | ((tdim1 & 0xFFFFu) << 16)),
               (int)((tdim1 >> 16) | (32u << 16)),           // tile_dim0 = 32
               tile1w4,                                      // tile_dim1 | tile_dim2<<16
               (int)I_,                                      // tensor_dim0_stride
               (int)((unsigned)(d1s & 0xFFFFull) << 16),     // dim1_stride[15:0]
               (int)(d1s >> 16) };                           // dim1_stride[47:16]
  i32x4 g2 = { tdim2, 0, 0, 0 };                             // tensor_dim2
  i32x4 g3 = { 0, 0, 0, 0 };
#if __clang_major__ >= 23
  i32x8 g4 = { 0, 0, 0, 0, 0, 0, 0, 0 };
  __builtin_amdgcn_tensor_load_to_lds(g0, g1, g2, g3, g4, 0);
#else
  __builtin_amdgcn_tensor_load_to_lds(g0, g1, g2, g3, 0);
#endif
#else
  (void)lds; (void)ga; (void)tdim1; (void)tile1w4; (void)d1s; (void)tdim2;
#endif
}

// ---------------------------------------------------------------------------
// Main GEMM (bf16 operands) + fused gate/LIF epilogue.
// ---------------------------------------------------------------------------
__global__ __launch_bounds__(256)
void snn_gemm_bf16(const unsigned short* __restrict__ Xb,  // [M, I] bf16
                   const unsigned short* __restrict__ Wb,  // [3H, I] bf16
                   const float* __restrict__ bih,
                   const float* __restrict__ bhh,
                   float* __restrict__ out) {
  __shared__ unsigned short As[2][TM * LDA];
  __shared__ unsigned short Bs[2][3 * TN * LDA];

  const int tid  = threadIdx.x;
  const int lane = tid & 31;
  const int wid  = tid >> 5;
  const int wm   = wid >> 1;     // 0..3 : 32-row slice
  const int wn   = wid & 1;      // 0..1 : 32-col slice
  const int l    = lane & 15;
  const int hf   = lane >> 4;

  const int row0 = blockIdx.y * TM;
  const int c0   = blockIdx.x * TN;

  v8f accR[2][2], accZ[2][2], accN[2][2];
  const v8f vzero = {0.f, 0.f, 0.f, 0.f, 0.f, 0.f, 0.f, 0.f};
#pragma unroll
  for (int mi = 0; mi < 2; ++mi)
#pragma unroll
    for (int ni = 0; ni < 2; ++ni) {
      accR[mi][ni] = vzero; accZ[mi][ni] = vzero; accN[mi][ni] = vzero;
    }

  const int nk = I_ / BK;   // 64

#if defined(HAVE_TDM)
  const unsigned long long gaX = (unsigned long long)(size_t)Xb;
  const unsigned long long gaW = (unsigned long long)(size_t)Wb;
  // prologue: tile 0 -> buffer 0
  if (wid == 0)
    tdm_issue((unsigned)(size_t)&As[0][0], gaX + (size_t)row0 * I_ * 2,
              (unsigned)M_, TM, 0ull, 0);
  else if (wid == 1)
    tdm_issue((unsigned)(size_t)&Bs[0][0], gaW + (size_t)c0 * I_ * 2,
              (unsigned)H_, TN | (3 << 16), (unsigned long long)H_ * I_, 3);
#endif

  for (int it = 0; it < nk; ++it) {
#if defined(HAVE_TDM)
    const int sb = it & 1;
    if (wid < 2) {
      if (it + 1 < nk) {            // issue next tile, then wait for current
        const int k1 = (it + 1) * BK;
        if (wid == 0)
          tdm_issue((unsigned)(size_t)&As[sb ^ 1][0],
                    gaX + ((size_t)row0 * I_ + k1) * 2, (unsigned)M_, TM, 0ull, 0);
        else
          tdm_issue((unsigned)(size_t)&Bs[sb ^ 1][0],
                    gaW + ((size_t)c0 * I_ + k1) * 2, (unsigned)H_,
                    TN | (3 << 16), (unsigned long long)H_ * I_, 3);
        __builtin_amdgcn_s_wait_tensorcnt(1);   // in-order: current tile done
      } else {
        __builtin_amdgcn_s_wait_tensorcnt(0);
      }
    }
    __syncthreads();
#else
    const int sb = 0;
    __syncthreads();
    // synchronous bf16 staging fallback (b128 in, b128 to LDS)
    for (int j = tid; j < TM * 4; j += 256) {
      const int row = j >> 2, q = j & 3;
      *(uint4*)&As[0][row * LDA + q * 8] =
          *(const uint4*)&Xb[(size_t)(row0 + row) * I_ + it * BK + q * 8];
    }
    for (int j = tid; j < 3 * TN * 4; j += 256) {
      const int r = j >> 2, q = j & 3;
      const int grow = (r >> 6) * H_ + c0 + (r & 63);
      *(uint4*)&Bs[0][r * LDA + q * 8] =
          *(const uint4*)&Wb[(size_t)grow * I_ + it * BK + q * 8];
    }
    __syncthreads();
#endif

    // A fragments (lane=M; K halves {0..7,16..23} / {8..15,24..31})
    Frag a[2];
#pragma unroll
    for (int mi = 0; mi < 2; ++mi) {
      const unsigned short* p = &As[sb][(wm * 32 + mi * 16 + l) * LDA + hf * 8];
      a[mi].h.lo = *(const v8bf*)(p);
      a[mi].h.hi = *(const v8bf*)(p + 16);
    }
    // B fragments (lane=N; K halves {0..7,16..23} / {8..15,24..31}) + 12 WMMAs
#pragma unroll
    for (int g = 0; g < 3; ++g) {
#pragma unroll
      for (int ni = 0; ni < 2; ++ni) {
        Frag bfr;
        const unsigned short* p =
            &Bs[sb][(g * TN + wn * 32 + ni * 16 + l) * LDA + hf * 8];
        bfr.h.lo = *(const v8bf*)(p);
        bfr.h.hi = *(const v8bf*)(p + 16);
#pragma unroll
        for (int mi = 0; mi < 2; ++mi) {
          v8f* acc = (g == 0) ? &accR[mi][ni] : (g == 1) ? &accZ[mi][ni] : &accN[mi][ni];
          *acc = __builtin_amdgcn_wmma_f32_16x16x32_bf16(
                     false, a[mi].v, false, bfr.v, (short)0, *acc, false, false);
        }
      }
    }
#if defined(HAVE_TDM)
    __syncthreads();   // all reads of buf[sb] done before it is re-DMA'd
#endif
  }

  // fused epilogue: bias + heaviside gates + 4-step LIF, all in registers.
  // C layout: VGPR v -> row (v + 8*hf); lane l -> column. Each lane holds
  // 8 consecutive rows = 2 batches x 4 contiguous timesteps.
#pragma unroll
  for (int ni = 0; ni < 2; ++ni) {
    const int c   = c0 + wn * 32 + ni * 16 + l;
    const float bir = bih[c],          bhr = bhh[c];
    const float biz = bih[H_ + c],     bhz = bhh[H_ + c];
    const float bin = bih[2 * H_ + c], bhn = bhh[2 * H_ + c];
#pragma unroll
    for (int mi = 0; mi < 2; ++mi) {
      const int r0 = row0 + wm * 32 + mi * 16 + hf * 8;
      float cur[8];
#pragma unroll
      for (int v = 0; v < 8; ++v) {
        const float yr = accR[mi][ni][v] + bir + bhr;
        const float r  = (yr > 0.0f) ? 1.0f : 0.0f;
        const float yz = accZ[mi][ni][v] + biz + bhz;
        const float z  = (yz > 0.0f) ? 1.0f : 0.0f;
        const float yn = accN[mi][ni][v] + bin + r * bhn;
        const float n  = (yn > 0.0f) ? 1.0f : 0.0f;
        cur[v] = (1.0f - z) * n;
      }
#pragma unroll
      for (int g2 = 0; g2 < 2; ++g2) {
        const int b = (r0 >> 2) + g2;
        float mem = 0.0f;
        float s[4];
#pragma unroll
        for (int t = 0; t < 4; ++t) {
          const float reset = (mem > 1.0f) ? 1.0f : 0.0f;   // THRESHOLD = 1
          mem = 0.99f * mem + cur[g2 * 4 + t] - reset;      // BETA = 0.99
          s[t] = (mem > 1.0f) ? 1.0f : 0.0f;
        }
        *(float4*)(out + ((size_t)b * H_ + c) * T_) = make_float4(s[0], s[1], s[2], s[3]);
      }
    }
  }
}

// ---------------------------------------------------------------------------
// Fallback (ws too small): round-1 fully fused fp32->bf16-on-the-fly kernel.
// ---------------------------------------------------------------------------
__global__ __launch_bounds__(256)
void snn_fused_f32(const float* __restrict__ X, const float* __restrict__ Wih,
                   const float* __restrict__ bih, const float* __restrict__ bhh,
                   float* __restrict__ out) {
  __shared__ unsigned short As[TM * LDA];
  __shared__ unsigned short Bs[3 * TN * LDA];

  const int tid = threadIdx.x, lane = tid & 31, wid = tid >> 5;
  const int wm = wid >> 1, wn = wid & 1, l = lane & 15, hf = lane >> 4;
  const int b0 = blockIdx.y * (TM / T_);
  const int c0 = blockIdx.x * TN;

  v8f accR[2][2], accZ[2][2], accN[2][2];
  const v8f vzero = {0.f, 0.f, 0.f, 0.f, 0.f, 0.f, 0.f, 0.f};
#pragma unroll
  for (int mi = 0; mi < 2; ++mi)
#pragma unroll
    for (int ni = 0; ni < 2; ++ni) {
      accR[mi][ni] = vzero; accZ[mi][ni] = vzero; accN[mi][ni] = vzero;
    }

  for (int k0 = 0; k0 < I_; k0 += BK) {
    __syncthreads();
#pragma unroll
    for (int j = 0; j < 4; ++j) {
      const int idx = tid + j * 256, bl = idx >> 5, kk = idx & 31;
      const float4 q = *(const float4*)(X + (size_t)(b0 + bl) * (I_ * T_)
                                          + (size_t)(k0 + kk) * T_);
      const int r = (bl * T_) * LDA + kk;
      As[r] = f2bf(q.x); As[r + LDA] = f2bf(q.y);
      As[r + 2 * LDA] = f2bf(q.z); As[r + 3 * LDA] = f2bf(q.w);
    }
#pragma unroll
    for (int j = 0; j < 6; ++j) {
      const int idx = tid + j * 256, na = idx >> 3, kq = idx & 7;
      const float4 q = *(const float4*)(Wih + (size_t)((na >> 6) * H_ + c0 + (na & 63)) * I_
                                            + k0 + kq * 4);
      const unsigned s0 = f2bf(q.x), s1 = f2bf(q.y), s2 = f2bf(q.z), s3 = f2bf(q.w);
      *(uint2*)&Bs[na * LDA + kq * 4] = make_uint2(s0 | (s1 << 16), s2 | (s3 << 16));
    }
    __syncthreads();

    Frag a[2];
#pragma unroll
    for (int mi = 0; mi < 2; ++mi) {
      const unsigned short* p = &As[(wm * 32 + mi * 16 + l) * LDA + hf * 8];
      a[mi].h.lo = *(const v8bf*)(p); a[mi].h.hi = *(const v8bf*)(p + 16);
    }
#pragma unroll
    for (int g = 0; g < 3; ++g)
#pragma unroll
      for (int ni = 0; ni < 2; ++ni) {
        Frag bfr;
        const unsigned short* p = &Bs[(g * TN + wn * 32 + ni * 16 + l) * LDA + hf * 8];
        bfr.h.lo = *(const v8bf*)(p); bfr.h.hi = *(const v8bf*)(p + 16);
#pragma unroll
        for (int mi = 0; mi < 2; ++mi) {
          v8f* acc = (g == 0) ? &accR[mi][ni] : (g == 1) ? &accZ[mi][ni] : &accN[mi][ni];
          *acc = __builtin_amdgcn_wmma_f32_16x16x32_bf16(
                     false, a[mi].v, false, bfr.v, (short)0, *acc, false, false);
        }
      }
  }

#pragma unroll
  for (int ni = 0; ni < 2; ++ni) {
    const int c = c0 + wn * 32 + ni * 16 + l;
    const float bir = bih[c], bhr = bhh[c];
    const float biz = bih[H_ + c], bhz = bhh[H_ + c];
    const float bin = bih[2 * H_ + c], bhn = bhh[2 * H_ + c];
#pragma unroll
    for (int mi = 0; mi < 2; ++mi) {
      const int r0 = blockIdx.y * TM + wm * 32 + mi * 16 + hf * 8;
      float cur[8];
#pragma unroll
      for (int v = 0; v < 8; ++v) {
        const float yr = accR[mi][ni][v] + bir + bhr;
        const float r  = (yr > 0.0f) ? 1.0f : 0.0f;
        const float yz = accZ[mi][ni][v] + biz + bhz;
        const float z  = (yz > 0.0f) ? 1.0f : 0.0f;
        const float yn = accN[mi][ni][v] + bin + r * bhn;
        const float n  = (yn > 0.0f) ? 1.0f : 0.0f;
        cur[v] = (1.0f - z) * n;
      }
#pragma unroll
      for (int g2 = 0; g2 < 2; ++g2) {
        const int b = (r0 >> 2) + g2;
        float mem = 0.0f, s[4];
#pragma unroll
        for (int t = 0; t < 4; ++t) {
          const float reset = (mem > 1.0f) ? 1.0f : 0.0f;
          mem = 0.99f * mem + cur[g2 * 4 + t] - reset;
          s[t] = (mem > 1.0f) ? 1.0f : 0.0f;
        }
        *(float4*)(out + ((size_t)b * H_ + c) * T_) = make_float4(s[0], s[1], s[2], s[3]);
      }
    }
  }
}

extern "C" void kernel_launch(void* const* d_in, const int* in_sizes, int n_in,
                              void* d_out, int out_size, void* d_ws, size_t ws_size,
                              hipStream_t stream) {
  (void)in_sizes; (void)n_in; (void)out_size;
  const float* X   = (const float*)d_in[0];   // inputs [B, I, T]
  const float* Wih = (const float*)d_in[1];   // [3H, I]
  const float* bih = (const float*)d_in[2];   // [3H]
  // d_in[3] = W_hh: dead (h0 == 0)
  const float* bhh = (const float*)d_in[4];   // [3H]
  float* out = (float*)d_out;                 // [B, H, T]

  const dim3 grid(H_ / TN, M_ / TM);          // (32, 64)

  if (ws_size >= WS_NEED) {
    unsigned short* Xb = (unsigned short*)d_ws;
    unsigned short* Wb = (unsigned short*)((char*)d_ws + XB_BYTES);
    cvt_x_bf16<<<(B_ * I_ / 4) / 256, 256, 0, stream>>>(X, Xb);
    cvt_w_bf16<<<(3 * H_ * I_ / 4) / 256, 256, 0, stream>>>(Wih, Wb);
    snn_gemm_bf16<<<grid, 256, 0, stream>>>(Xb, Wb, bih, bhh, out);
  } else {
    snn_fused_f32<<<grid, 256, 0, stream>>>(X, Wih, bih, bhh, out);
  }
}
